// MambaITD_28509992911078
// MI455X (gfx1250) — matrix operand order repaired and checked by reading, not compiled, not run on hardware
//
#include <hip/hip_runtime.h>
#include <hip/hip_bf16.h>
#include <math.h>
#include <stdint.h>

// ---------------------------------------------------------------------------
// MambaITD forward for MI455X (gfx1250, wave32, WMMA + TDM)
// Shapes: B=8, L=256 -> M=2048 tokens, D=256, DI=512, DS=16, DTR=16, K=4,
// NL=4 layers x 2 branches, FS=64, HID=512, V=5000.
// ---------------------------------------------------------------------------

#define Mtok   2048      // B*L
#define Ddim   256
#define DIdim  512
#define DSdim  16
#define DTRdim 16
#define Kconv  4
#define Bdim   8
#define Ldim   256
#define FSdim  64
#define HIDdim 512

typedef __attribute__((ext_vector_type(16))) _Float16     v16h;
typedef __attribute__((ext_vector_type(8)))  float        v8f;
typedef __attribute__((ext_vector_type(2)))  float        v2f;
typedef __attribute__((ext_vector_type(4)))  unsigned int u32x4;
typedef __attribute__((ext_vector_type(8)))  int          i32x8;
typedef __attribute__((ext_vector_type(4)))  int          i32x4;

#if defined(__has_builtin)
#  if __has_builtin(__builtin_amdgcn_tensor_load_to_lds) && \
      __has_builtin(__builtin_amdgcn_s_wait_tensorcnt)
#    define HAVE_TDM 1
#  endif
#endif
#ifndef HAVE_TDM
#  define HAVE_TDM 0
#endif

__device__ __forceinline__ float wave_reduce_sum(float v) {
#pragma unroll
  for (int off = 16; off > 0; off >>= 1)
    v += __shfl_xor(v, off, 32);
  return v;
}

__device__ __forceinline__ float sigmoidf_fast(float x) {
  return 1.0f / (1.0f + __expf(-x));
}

// Low 32 bits of a generic pointer to __shared__ == byte offset in LDS space.
__device__ __forceinline__ unsigned lds_byte_off(const void* p) {
  return (unsigned)(uintptr_t)p;
}

// ---------------------------------------------------------------------------
// TDM: DMA a 2-D weight tile  (rows x 32 f32, row stride Kd f32)  into LDS.
// LDS layout: padded rows of 36 f32 (pad 4 DWORDs every 32 -> conflict-free
// 16-row B-fragment reads).  Tracked on TENSORcnt.
// D# group0: count=1 | lds_addr | global_addr[56:0] | type=2
// D# group1: data_size=4B, pad_enable, pad_interval=32DW(code 4),
//            pad_amount=4DW(code 3); tensor dims == tile dims (no OOB);
//            tensor_dim0_stride = Kd.   Groups 2/3 zero (2-D tensor).
// ---------------------------------------------------------------------------
__device__ __forceinline__ void tdm_load_tile(unsigned lds_addr,
                                              const float* gptr,
                                              int rows, int Kd)
{
#if HAVE_TDM
  unsigned long long ga = (unsigned long long)(uintptr_t)gptr;
  u32x4 g0;
  g0[0] = 1u;                                        // count=1 (user D#)
  g0[1] = lds_addr;                                  // LDS dest (bytes)
  g0[2] = (unsigned)(ga & 0xffffffffu);              // global_addr[31:0]
  g0[3] = (unsigned)((ga >> 32) & 0x01ffffffu)       // global_addr[56:32]
        | (2u << 30);                                // type = 2 ("image")

  i32x8 g1;
  g1[0] = (2 << 16)                                  // data_size = 4 bytes
        | (1 << 20)                                  // pad_enable
        | (4 << 22)                                  // pad_interval: 32 DWORDs
        | (3 << 25);                                 // pad_amount:   4 DWORDs
  g1[1] = (int)(32u << 16);                          // tensor_dim0 = 32 (lo16)
  g1[2] = (int)((unsigned)rows << 16);               // dim0 hi=0 | tensor_dim1 lo16
  g1[3] = (int)(32u << 16);                          // dim1 hi=0 | tile_dim0 = 32
  g1[4] = rows;                                      // tile_dim1 | tile_dim2=0
  g1[5] = Kd;                                        // tensor_dim0_stride lo32
  g1[6] = 0;                                         // stride hi | dim1_stride lo
  g1[7] = 0;

  i32x4 z4 = {};
#  if __has_include(<hip/amd_detail/amd_gfx1250_TDM.h>)
  i32x8 z8 = {};
  __builtin_amdgcn_tensor_load_to_lds(g0, g1, z4, z4, z8, 0);
#  else
  __builtin_amdgcn_tensor_load_to_lds(g0, g1, z4, z4, 0);
#  endif
#else
  (void)lds_addr; (void)gptr; (void)rows; (void)Kd;
#endif
}

// ---------------------------------------------------------------------------
// WMMA GEMM:  C[M,N] = A[M,K] * W[N,K]^T   (f32 in, f16 mul, f32 acc)
// Wave tile 16 x (16*NT); block = 4 waves stacked over M -> 64 x (16*NT).
// Weight tile (16*NT rows x 32 K) staged in LDS by the TDM, double-buffered
// on TENSORcnt; all 4 waves consume one copy.  A-fragments via global loads.
// grid = (N/(16*NT), M/64).  EPI: 0 store, 1 +=Res, 2 relu(.+Bias)
// ---------------------------------------------------------------------------
template<int NT, int EPI>
__global__ __launch_bounds__(128)
void gemm_wmma_kernel(const float* __restrict__ A, const float* __restrict__ W,
                      float* __restrict__ C, const float* __restrict__ Res,
                      const float* __restrict__ Bias,
                      int M, int N, int Kd)
{
  constexpr int ROWS = 16 * NT;   // weight rows per block tile
  constexpr int LROW = 36;        // padded LDS row pitch (f32)
  __shared__ float ldsW[2][64 * LROW];

  const int lane  = threadIdx.x & 31;
  const int wave  = threadIdx.x >> 5;
  const int half  = lane >> 4;
  const int l16   = lane & 15;
  const int tileN = blockIdx.x * ROWS;
  const int tileM = (blockIdx.y * 4 + wave) * 16;

  const float* __restrict__ arow  = A + (size_t)(tileM + l16) * Kd;
  const float* __restrict__ wtile = W + (size_t)tileN * Kd;

  v8f acc[NT];
#pragma unroll
  for (int t = 0; t < NT; ++t) acc[t] = (v8f){};

  int cur = 0;
#if HAVE_TDM
  if (wave == 0)
    tdm_load_tile(lds_byte_off(&ldsW[0][0]), wtile, ROWS, Kd);
#endif

  for (int k0 = 0; k0 < Kd; k0 += 32) {
#if HAVE_TDM
    if (wave == 0) {
      if (k0 + 32 < Kd) {
        tdm_load_tile(lds_byte_off(&ldsW[cur ^ 1][0]), wtile + k0 + 32, ROWS, Kd);
        // in-order TENSORcnt: <=1 outstanding  =>  tile[cur] complete
        __builtin_amdgcn_s_wait_tensorcnt(1);
      } else {
        __builtin_amdgcn_s_wait_tensorcnt(0);   // drain: last tile complete
      }
    }
    __syncthreads();
#else
    // cooperative fallback copy (same padded layout)
    __syncthreads();
    for (int e = threadIdx.x; e < ROWS * 32; e += 128) {
      int r = e >> 5, c = e & 31;
      ldsW[cur][r * LROW + c] = wtile[(size_t)r * Kd + k0 + c];
    }
    __syncthreads();
#endif
    const float* lw = ldsW[cur];

    // A fragment (16-bit A 16x32; K order is this code's, not ISA table): lane row m = tileM+l16,
    // contiguous K run [half*16, +16) (matches B fragment).
    v16h a;
#pragma unroll
    for (int i = 0; i < 8; ++i) {
      a[i]     = (_Float16)arow[k0 + (half << 4) + i];
      a[i + 8] = (_Float16)arow[k0 + (half << 4) + 8 + i];
    }
#pragma unroll
    for (int t = 0; t < NT; ++t) {
      // B fragment from LDS: col n = tileN+t*16+l16, K run [half*16, +16)
      const float* lrow = lw + (t * 16 + l16) * LROW + (half << 4);
      v16h b;
#pragma unroll
      for (int i = 0; i < 16; ++i)
        b[i] = (_Float16)lrow[i];
      acc[t] = __builtin_amdgcn_wmma_f32_16x16x32_f16(
          false, a, false, b, (short)0, acc[t], false, false);
    }
    __syncthreads();   // readers done before this buffer is re-filled
    cur ^= 1;
  }

#pragma unroll
  for (int t = 0; t < NT; ++t) {
#pragma unroll
    for (int r = 0; r < 8; ++r) {
      int m = tileM + r + (half << 3);
      int n = tileN + t * 16 + l16;
      size_t idx = (size_t)m * N + n;
      float v = acc[t][r];
      if (EPI == 1) v += Res[idx];
      if (EPI == 2) v = fmaxf(v + Bias[n], 0.0f);
      C[idx] = v;
    }
  }
}

// ---------------------------------------------------------------------------
// dtproj via fp32 WMMA (V_WMMA_F32_16X16X4_F32), K=16 fully unrolled.
// dt[m,n] = softplus( dbc[m,0:16] . Wt[n,:] + Bias[n] ),  n < 512.
// grid = (512/64, 2048/64), block = 128.
// ---------------------------------------------------------------------------
__global__ __launch_bounds__(128)
void dtproj_wmma_kernel(const float* __restrict__ dbc,   // (Mtok, 48)
                        const float* __restrict__ Wt,    // (512, 16)
                        const float* __restrict__ Bias,  // (512)
                        float* __restrict__ dt)          // (Mtok, 512)
{
  const int lane  = threadIdx.x & 31;
  const int wave  = threadIdx.x >> 5;
  const int half  = lane >> 4;
  const int l16   = lane & 15;
  const int tileN = blockIdx.x * 64;
  const int tileM = (blockIdx.y * 4 + wave) * 16;

  const float* __restrict__ arow = dbc + (size_t)(tileM + l16) * 48;

  v8f acc[4];
#pragma unroll
  for (int t = 0; t < 4; ++t) acc[t] = (v8f){};

#pragma unroll
  for (int k0 = 0; k0 < DTRdim; k0 += 4) {
    v2f a;
    a[0] = arow[k0 + (half << 1) + 0];
    a[1] = arow[k0 + (half << 1) + 1];
#pragma unroll
    for (int t = 0; t < 4; ++t) {
      const float* __restrict__ wrow = Wt + (size_t)(tileN + t * 16 + l16) * DTRdim;
      v2f b;
      b[0] = wrow[k0 + (half << 1) + 0];
      b[1] = wrow[k0 + (half << 1) + 1];
      acc[t] = __builtin_amdgcn_wmma_f32_16x16x4_f32(
          false, a, false, b, (short)0, acc[t], false, false);
    }
  }

#pragma unroll
  for (int t = 0; t < 4; ++t) {
#pragma unroll
    for (int r = 0; r < 8; ++r) {
      int m = tileM + r + (half << 3);
      int n = tileN + t * 16 + l16;
      float s = acc[t][r] + Bias[n];
      dt[(size_t)m * DIdim + n] = (s > 20.0f) ? s : log1pf(__expf(s));
    }
  }
}

// ---------------------------------------------------------------------------
// Embedding + continuous-branch input:  Xb = embed[S_b], Xc = log(S_c+eps)*wi
// ---------------------------------------------------------------------------
__global__ void embed_kernel(const int* __restrict__ Sb, const float* __restrict__ Sc,
                             const float* __restrict__ embed,
                             const float* __restrict__ wi_w, const float* __restrict__ wi_b,
                             float* __restrict__ Xb, float* __restrict__ Xc)
{
  int idx = blockIdx.x * blockDim.x + threadIdx.x;
  if (idx >= Mtok * Ddim) return;
  int row = idx >> 8;           // Ddim = 256
  int d   = idx & 255;
  Xb[idx] = embed[(size_t)Sb[row] * Ddim + d];
  Xc[idx] = logf(Sc[row] + 1e-8f) * wi_w[d] + wi_b[d];
}

// e_x[b,d] = X[b,:] . ws_w[d,:] + ws_b[d]    (K = 64, tiny)
__global__ void ex_kernel(const float* __restrict__ X, const float* __restrict__ ws_w,
                          const float* __restrict__ ws_b, float* __restrict__ ex)
{
  int idx = blockIdx.x * blockDim.x + threadIdx.x;
  if (idx >= Bdim * Ddim) return;
  int b = idx >> 8, d = idx & 255;
  float s = ws_b[d];
#pragma unroll 8
  for (int f = 0; f < FSdim; ++f) s += X[b * FSdim + f] * ws_w[d * FSdim + f];
  ex[idx] = s;
}

// ---------------------------------------------------------------------------
// LayerNorm: one wave per row of 256.  grid = Mtok/8 blocks of 256 threads.
// ---------------------------------------------------------------------------
__global__ __launch_bounds__(256)
void ln_kernel(const float* __restrict__ X, const float* __restrict__ w,
               const float* __restrict__ b, float* __restrict__ Y)
{
  int wave = threadIdx.x >> 5;
  int lane = threadIdx.x & 31;
  int row  = blockIdx.x * 8 + wave;
  const float* xr = X + (size_t)row * Ddim;
  float v[8]; float s = 0.f;
#pragma unroll
  for (int i = 0; i < 8; ++i) { v[i] = xr[lane + 32 * i]; s += v[i]; }
  s = wave_reduce_sum(s);
  float mean = s * (1.0f / Ddim);
  float q = 0.f;
#pragma unroll
  for (int i = 0; i < 8; ++i) { float d = v[i] - mean; q += d * d; }
  q = wave_reduce_sum(q);
  float rs = rsqrtf(q * (1.0f / Ddim) + 1e-5f);
#pragma unroll
  for (int i = 0; i < 8; ++i) {
    int d = lane + 32 * i;
    Y[(size_t)row * Ddim + d] = (v[i] - mean) * rs * w[d] + b[d];
  }
}

// ---------------------------------------------------------------------------
// Causal depthwise conv (K=4) over u = xz[:, :DI], then SiLU.
// ---------------------------------------------------------------------------
__global__ void conv_silu_kernel(const float* __restrict__ xz, const float* __restrict__ cw,
                                 const float* __restrict__ cb, float* __restrict__ u)
{
  int idx = blockIdx.x * blockDim.x + threadIdx.x;
  if (idx >= Mtok * DIdim) return;
  int c   = idx & 511;
  int row = idx >> 9;
  int l   = row & 255;
  float s = cb[c];
#pragma unroll
  for (int k = 0; k < Kconv; ++k) {
    int lk = l - 3 + k;
    if (lk >= 0)
      s += xz[(size_t)(row - 3 + k) * (2 * DIdim) + c] * cw[c * Kconv + k];
  }
  u[idx] = s * sigmoidf_fast(s);
}

// ---------------------------------------------------------------------------
// Selective scan: sequential over L, parallel over (b, channel).
// One thread per (b,d): 16-element state in registers.  Fuses +Dp*u and
// *silu(z) epilogue.   4096 threads.
// ---------------------------------------------------------------------------
__global__ __launch_bounds__(256)
void scan_kernel(const float* __restrict__ dt, const float* __restrict__ u,
                 const float* __restrict__ xz,   // z = xz[:, DI:2DI]
                 const float* __restrict__ dbc,  // B = [16:32), C = [32:48)
                 const float* __restrict__ A_log, const float* __restrict__ Dp,
                 float* __restrict__ yz)
{
  int g = blockIdx.x * blockDim.x + threadIdx.x;
  if (g >= Bdim * DIdim) return;
  int b = g >> 9;
  int d = g & 511;

  float Ac[DSdim], h[DSdim];
#pragma unroll
  for (int s = 0; s < DSdim; ++s) {
    Ac[s] = -__expf(A_log[d * DSdim + s]);
    h[s]  = 0.f;
  }
  float dpc = Dp[d];

  for (int l = 0; l < Ldim; ++l) {
    int row = b * Ldim + l;
    float dtv = dt[(size_t)row * DIdim + d];
    float uv  = u[(size_t)row * DIdim + d];
    float zv  = xz[(size_t)row * (2 * DIdim) + DIdim + d];
    const float* bc = dbc + row * 48;
    float y = 0.f;
#pragma unroll
    for (int s = 0; s < DSdim; ++s) {
      float dA = __expf(dtv * Ac[s]);
      h[s] = dA * h[s] + dtv * bc[16 + s] * uv;
      y   += h[s] * bc[32 + s];
    }
    y = (y + dpc * uv) * (zv * sigmoidf_fast(zv));
    yz[(size_t)row * DIdim + d] = y;
  }
}

// G[b,d] = sigmoid( e_x[b,:] . gate_w[d,:] + gate_b[d] )
__global__ void gate_kernel(const float* __restrict__ ex, const float* __restrict__ gw,
                            const float* __restrict__ gb, float* __restrict__ G)
{
  int idx = blockIdx.x * blockDim.x + threadIdx.x;
  if (idx >= Bdim * Ddim) return;
  int b = idx >> 8, d = idx & 255;
  float s = gb[d];
#pragma unroll 8
  for (int f = 0; f < Ddim; ++f) s += ex[b * Ddim + f] * gw[d * Ddim + f];
  G[idx] = sigmoidf_fast(s);
}

// F_fus = LN(G*Hb + (1-G)*Hc); F_fin = concat(F_fus, e_x broadcast)
__global__ __launch_bounds__(256)
void fuse_kernel(const float* __restrict__ G, const float* __restrict__ Hb,
                 const float* __restrict__ Hc, const float* __restrict__ lnw,
                 const float* __restrict__ lnb, const float* __restrict__ ex,
                 float* __restrict__ F)
{
  int wave = threadIdx.x >> 5;
  int lane = threadIdx.x & 31;
  int row  = blockIdx.x * 8 + wave;
  int b    = row >> 8;   // Ldim = 256
  float v[8]; float s = 0.f;
#pragma unroll
  for (int i = 0; i < 8; ++i) {
    int d = lane + 32 * i;
    float g = G[b * Ddim + d];
    float x = g * Hb[(size_t)row * Ddim + d] + (1.f - g) * Hc[(size_t)row * Ddim + d];
    v[i] = x; s += x;
  }
  s = wave_reduce_sum(s);
  float mean = s * (1.0f / Ddim);
  float q = 0.f;
#pragma unroll
  for (int i = 0; i < 8; ++i) { float d = v[i] - mean; q += d * d; }
  q = wave_reduce_sum(q);
  float rs = rsqrtf(q * (1.0f / Ddim) + 1e-5f);
#pragma unroll
  for (int i = 0; i < 8; ++i) {
    int d = lane + 32 * i;
    F[(size_t)row * (2 * Ddim) + d]        = (v[i] - mean) * rs * lnw[d] + lnb[d];
    F[(size_t)row * (2 * Ddim) + Ddim + d] = ex[b * Ddim + d];
  }
}

// out[row] = Hh[row,:] . fc2_w + fc2_b[0]    one wave per row
__global__ __launch_bounds__(256)
void fc2_kernel(const float* __restrict__ Hh, const float* __restrict__ w2,
                const float* __restrict__ b2, float* __restrict__ out)
{
  int wave = threadIdx.x >> 5;
  int lane = threadIdx.x & 31;
  int row  = blockIdx.x * 8 + wave;
  const float* hr = Hh + (size_t)row * HIDdim;
  float s = 0.f;
#pragma unroll
  for (int i = 0; i < 16; ++i) s += hr[lane + 32 * i] * w2[lane + 32 * i];
  s = wave_reduce_sum(s);
  if (lane == 0) out[row] = s + b2[0];
}

// ---------------------------------------------------------------------------
// Host-side pipeline
// ---------------------------------------------------------------------------
static inline void launch_gemm(const float* A, const float* W, float* C,
                               const float* Res, const float* Bias,
                               int M, int N, int Kd, int epi, hipStream_t s)
{
  const float* R = Res  ? Res  : C;
  const float* B = Bias ? Bias : C;
  if (N % 64 == 0) {
    dim3 grid(N / 64, M / 64);
    switch (epi) {
      case 0: gemm_wmma_kernel<4,0><<<grid,128,0,s>>>(A,W,C,R,B,M,N,Kd); break;
      case 1: gemm_wmma_kernel<4,1><<<grid,128,0,s>>>(A,W,C,R,B,M,N,Kd); break;
      default:gemm_wmma_kernel<4,2><<<grid,128,0,s>>>(A,W,C,R,B,M,N,Kd); break;
    }
  } else {               // N == 48 (xproj)
    dim3 grid(N / 48, M / 64);
    gemm_wmma_kernel<3,0><<<grid,128,0,s>>>(A,W,C,R,B,M,N,Kd);
  }
}

extern "C" void kernel_launch(void* const* d_in, const int* in_sizes, int n_in,
                              void* d_out, int out_size, void* d_ws, size_t ws_size,
                              hipStream_t stream)
{
  // ---- inputs (setup_inputs dict order) ----
  const int*   S_b   = (const int*)  d_in[0];
  const float* S_c   = (const float*)d_in[1];
  const float* Xin   = (const float*)d_in[2];
  const float* embed = (const float*)d_in[3];
  const float* wi_w  = (const float*)d_in[4];
  const float* wi_b  = (const float*)d_in[5];
  const float* ws_w  = (const float*)d_in[6];
  const float* ws_b  = (const float*)d_in[7];
  // branch params: base index 8 ('b') and 19 ('c'); order within branch:
  // ln_w, ln_b, inproj_w, conv_w, conv_b, xproj_w, dtproj_w, dtproj_b, A_log, Dp, outproj_w
  const float* gate_w = (const float*)d_in[30];
  const float* gate_b = (const float*)d_in[31];
  const float* lnf_w  = (const float*)d_in[32];
  const float* lnf_b  = (const float*)d_in[33];
  const float* fc1_w  = (const float*)d_in[34];
  const float* fc1_b  = (const float*)d_in[35];
  const float* fc2_w  = (const float*)d_in[36];
  const float* fc2_b  = (const float*)d_in[37];
  float* out = (float*)d_out;

  // ---- workspace carve-up (floats) ----
  float* ws = (float*)d_ws;
  float* Xb   = ws;                      // 2048*256
  float* Xc   = Xb  + Mtok * Ddim;       // 2048*256
  float* h    = Xc  + Mtok * Ddim;       // 2048*256
  float* xz   = h   + Mtok * Ddim;       // 2048*1024
  float* u    = xz  + Mtok * 2 * DIdim;  // 2048*512
  float* dbc  = u   + Mtok * DIdim;      // 2048*48
  float* dtb  = dbc + Mtok * 48;         // 2048*512
  float* yzb  = dtb + Mtok * DIdim;      // 2048*512
  float* F    = yzb + Mtok * DIdim;      // 2048*512
  float* Hh   = F   + Mtok * 2 * Ddim;   // 2048*512
  float* ex   = Hh  + Mtok * HIDdim;     // 2048
  float* G    = ex  + Bdim * Ddim;       // 2048

  const int T = 256;

  // 1. embeddings + e_x
  embed_kernel<<<(Mtok * Ddim + T - 1) / T, T, 0, stream>>>(
      S_b, S_c, embed, wi_w, wi_b, Xb, Xc);
  ex_kernel<<<(Bdim * Ddim + T - 1) / T, T, 0, stream>>>(Xin, ws_w, ws_b, ex);

  // 2. two Mamba encoder branches
  for (int br = 0; br < 2; ++br) {
    int p = (br == 0) ? 8 : 19;
    const float* ln_w    = (const float*)d_in[p + 0];
    const float* ln_b    = (const float*)d_in[p + 1];
    const float* inproj  = (const float*)d_in[p + 2];
    const float* conv_w  = (const float*)d_in[p + 3];
    const float* conv_b  = (const float*)d_in[p + 4];
    const float* xproj   = (const float*)d_in[p + 5];
    const float* dtw     = (const float*)d_in[p + 6];
    const float* dtbias  = (const float*)d_in[p + 7];
    const float* A_log   = (const float*)d_in[p + 8];
    const float* Dp      = (const float*)d_in[p + 9];
    const float* outproj = (const float*)d_in[p + 10];
    float* X = (br == 0) ? Xb : Xc;

    for (int i = 0; i < 4; ++i) {
      ln_kernel<<<Mtok / 8, 256, 0, stream>>>(X, ln_w + i * Ddim, ln_b + i * Ddim, h);
      // xz = h @ inproj^T          (2048 x 1024, K=256)
      launch_gemm(h, inproj + (size_t)i * 2 * DIdim * Ddim, xz,
                  nullptr, nullptr, Mtok, 2 * DIdim, Ddim, 0, stream);
      conv_silu_kernel<<<(Mtok * DIdim + T - 1) / T, T, 0, stream>>>(
          xz, conv_w + i * DIdim * Kconv, conv_b + i * DIdim, u);
      // dbc = u @ xproj^T          (2048 x 48, K=512)
      launch_gemm(u, xproj + (size_t)i * 48 * DIdim, dbc,
                  nullptr, nullptr, Mtok, 48, DIdim, 0, stream);
      // dt = softplus(dbc[:, :16] @ dtw^T + dtb)   fp32 WMMA 16x16x4
      {
        dim3 grid(DIdim / 64, Mtok / 64);
        dtproj_wmma_kernel<<<grid, 128, 0, stream>>>(
            dbc, dtw + (size_t)i * DIdim * DTRdim, dtbias + i * DIdim, dtb);
      }
      scan_kernel<<<(Bdim * DIdim + T - 1) / T, T, 0, stream>>>(
          dtb, u, xz, dbc, A_log + (size_t)i * DIdim * DSdim, Dp + i * DIdim, yzb);
      // X = X + yz @ outproj^T     (2048 x 256, K=512, fused residual)
      launch_gemm(yzb, outproj + (size_t)i * Ddim * DIdim, X,
                  X, nullptr, Mtok, Ddim, DIdim, 1, stream);
    }
  }

  // 3. gate + fusion + LN + concat
  gate_kernel<<<(Bdim * Ddim + T - 1) / T, T, 0, stream>>>(ex, gate_w, gate_b, G);
  fuse_kernel<<<Mtok / 8, 256, 0, stream>>>(G, Xb, Xc, lnf_w, lnf_b, ex, F);

  // 4. head: Hh = relu(F @ fc1^T + b1) ; out = Hh @ fc2^T + b2
  launch_gemm(F, fc1_w, Hh, nullptr, fc1_b, Mtok, HIDdim, 2 * Ddim, 2, stream);
  fc2_kernel<<<Mtok / 8, 256, 0, stream>>>(Hh, fc2_w, fc2_b, out);
}
